// AdaptiveSoftmaxRNN_18786186953329
// MI455X (gfx1250) — matrix operand order repaired and checked by reading, not compiled, not run on hardware
//
#include <hip/hip_runtime.h>
#include <hip/hip_bf16.h>

// ---------------------------------------------------------------------------
// AdaptiveSoftmaxRNN for MI455X (gfx1250): bf16 WMMA + Tensor Data Mover.
//   - weights cast fp32->bf16 once per call
//   - LSTM input projections hoisted into big WMMA GEMMs
//   - recurrent h@W_hh^T done per-step with 4-gate-fused WMMA blocks
//   - adaptive softmax via streaming tile-wise logsumexp (never materializes
//     the full [4096 x 20002] logits)
//   - LDS tile staging via TENSOR_LOAD_TO_LDS (TDM) where the toolchain
//     exposes the builtin; synchronous vmem->LDS fallback otherwise.
// Workspace use: ~260 MB.
// ---------------------------------------------------------------------------

typedef __bf16 bf16_t;
typedef bf16_t v16bf __attribute__((ext_vector_type(16)));
typedef float  v8f   __attribute__((ext_vector_type(8)));

#define SEQ    128
#define BATCH  32
#define NROWS  4096            // SEQ*BATCH
#define NINP   1024
#define NHID   1024
#define GDIM   4096            // 4*NHID
#define HEADN  20002
#define SL     20000
#define CUT0   20000
#define CUT1   50000
#define T0HS   512
#define T0OS   30000
#define T1HS   256
#define T1OS   50000
#define LTILE  128
#define NT_HEAD ((HEADN + LTILE - 1) / LTILE)   // 157
#define NT_T0   ((T0OS  + LTILE - 1) / LTILE)   // 235
#define NT_T1   ((T1OS  + LTILE - 1) / LTILE)   // 391

// ---- TDM availability (device pass only; host pass parses the fallback) ----
#if defined(__HIP_DEVICE_COMPILE__) && defined(__has_builtin)
#  if __has_builtin(__builtin_amdgcn_tensor_load_to_lds)
#    define HAVE_TDM 1
#  endif
#endif
#ifndef HAVE_TDM
#  define HAVE_TDM 0
#endif

#if HAVE_TDM
typedef unsigned int u32x4 __attribute__((ext_vector_type(4)));
typedef int          i32x4 __attribute__((ext_vector_type(4)));
typedef int          i32x8 __attribute__((ext_vector_type(8)));

__device__ __forceinline__ unsigned lds_off_u32(const void* p) {
  return (unsigned)(unsigned long long)
      (__attribute__((address_space(3))) const void*)p;
}

// 2D bf16 tile DMA: rows x cols elements from row-major [.. , stride] matrix
// into LDS (packed, row-major, cols*2 bytes per row). Rows beyond tensor_rows
// are zero-filled by TDM OOB handling.
__device__ __forceinline__ void tdm_load_2d_bf16(const unsigned short* gsrc,
                                                 unsigned lds_off,
                                                 unsigned rows, unsigned cols,
                                                 unsigned stride_elems,
                                                 unsigned tensor_rows) {
  unsigned long long ga = (unsigned long long)(size_t)gsrc;
  u32x4 g0;
  g0.x = 1u;                                          // count=1 (valid), user
  g0.y = lds_off;                                     // lds_addr
  g0.z = (unsigned)(ga & 0xFFFFFFFFu);                // global_addr[31:0]
  g0.w = (unsigned)((ga >> 32) & 0x1FFFFFFu)          // global_addr[56:32]
         | (2u << 30);                                // type = 2 ("image")
  unsigned td0 = cols;                                // tensor_dim0 (elems)
  unsigned td1 = tensor_rows;                         // tensor_dim1 (rows)
  i32x8 g1;
  g1[0] = (int)(1u << 16);                            // data_size=1 (2 bytes)
  g1[1] = (int)((td0 & 0xFFFFu) << 16);               // dim0[15:0] @ bits63:48
  g1[2] = (int)((td0 >> 16) | ((td1 & 0xFFFFu) << 16));
  g1[3] = (int)((td1 >> 16) | ((cols & 0xFFFFu) << 16));   // tile_dim0
  g1[4] = (int)(rows & 0xFFFFu);                      // tile_dim1, tile_dim2=0
  g1[5] = (int)stride_elems;                          // dim0_stride[31:0]
  g1[6] = 0;                                          // stride[47:32], s1=0
  g1[7] = 0;
  i32x4 z4 = {0, 0, 0, 0};
#if __clang_major__ >= 23
  i32x8 z8 = {0, 0, 0, 0, 0, 0, 0, 0};
  __builtin_amdgcn_tensor_load_to_lds(g0, g1, z4, z4, z8, 0);
#else
  __builtin_amdgcn_tensor_load_to_lds(g0, g1, z4, z4, 0);
#endif
}

__device__ __forceinline__ void tdm_wait0() {
#if __has_builtin(__builtin_amdgcn_s_wait_tensorcnt)
  __builtin_amdgcn_s_wait_tensorcnt(0);
#else
  asm volatile("s_wait_tensorcnt 0x0" ::: "memory");
#endif
}
#endif  // HAVE_TDM

__device__ __forceinline__ unsigned short f2bf(float x) {
  unsigned int u = __float_as_uint(x);
  u += 0x7FFFu + ((u >> 16) & 1u);           // round-to-nearest-even
  return (unsigned short)(u >> 16);
}

union FragAB { uint4 q[2]; v16bf v; };

__device__ __forceinline__ v8f zero8() {
  v8f z = {0.f, 0.f, 0.f, 0.f, 0.f, 0.f, 0.f, 0.f};
  return z;
}

__device__ __forceinline__ v8f wmma_bf16(v16bf a, v16bf b, v8f c) {
  return __builtin_amdgcn_wmma_f32_16x16x32_bf16(false, a, false, b,
                                                 (short)0, c, false, false);
}

__device__ __forceinline__ float sigm(float x) {
  return 1.0f / (1.0f + __expf(-x));
}

// ---------------------------------------------------------------------------
// fp32 -> bf16 cast (n must be a multiple of 4)
// ---------------------------------------------------------------------------
__global__ void cast_f32_bf16_kernel(const float* __restrict__ in,
                                     unsigned short* __restrict__ out, long n) {
  long i = (long)blockIdx.x * blockDim.x + threadIdx.x;
  long idx = i * 4;
  if (idx + 3 < n) {
    float4 v = ((const float4*)in)[i];
    out[idx + 0] = f2bf(v.x);
    out[idx + 1] = f2bf(v.y);
    out[idx + 2] = f2bf(v.z);
    out[idx + 3] = f2bf(v.w);
  }
}

// ---------------------------------------------------------------------------
// embedding gather -> bf16 rows
// ---------------------------------------------------------------------------
__global__ void embed_kernel(const int* __restrict__ tokens,
                             const float* __restrict__ embW,
                             unsigned short* __restrict__ X) {
  int row = blockIdx.x;
  int tok = tokens[row];
  const float4* src = (const float4*)(embW + (size_t)tok * NINP);
  unsigned short* dst = X + (size_t)row * NINP;
  for (int i = threadIdx.x; i < NINP / 4; i += blockDim.x) {
    float4 v = src[i];
    dst[i * 4 + 0] = f2bf(v.x);
    dst[i * 4 + 1] = f2bf(v.y);
    dst[i * 4 + 2] = f2bf(v.z);
    dst[i * 4 + 3] = f2bf(v.w);
  }
}

// ---------------------------------------------------------------------------
// Generic WMMA GEMM: C[M,N] = A[M,K] * B[N,K]^T (+ bias1[n] + bias2[n])
// M,N multiples of 64; K multiple of 32. Block = 256 thr (8 waves),
// tile 64x64, each wave computes 16x32 (2 WMMA tiles).
// ---------------------------------------------------------------------------
template <bool OUT_BF16, bool BIAS>
__global__ void wmma_gemm_kernel(const unsigned short* __restrict__ A,
                                 const unsigned short* __restrict__ B,
                                 float* __restrict__ Cf,
                                 unsigned short* __restrict__ Cb,
                                 const float* __restrict__ bias1,
                                 const float* __restrict__ bias2,
                                 int M, int N, int K) {
  __shared__ __align__(16) unsigned short As[64 * 32];
  __shared__ __align__(16) unsigned short Bs[64 * 32];
  const int tid = threadIdx.x;
  const int lane = tid & 31, w = tid >> 5;
  const int wm = w & 3, wn = w >> 2;
  const int n0 = blockIdx.x * 64, m0 = blockIdx.y * 64;
  const int lr = lane & 15, lh = lane >> 4;
  v8f acc0 = zero8(), acc1 = zero8();

  for (int k0 = 0; k0 < K; k0 += 32) {
#if HAVE_TDM
    if (w == 0) {
      tdm_load_2d_bf16(A + (size_t)m0 * K + k0, lds_off_u32(As), 64, 32, K, 64);
      tdm_load_2d_bf16(B + (size_t)n0 * K + k0, lds_off_u32(Bs), 64, 32, K, 64);
      tdm_wait0();
    }
#else
    {
      const int r = tid >> 2, seg = tid & 3;
      *(uint4*)(As + r * 32 + seg * 8) =
          *(const uint4*)(A + (size_t)(m0 + r) * K + k0 + seg * 8);
      *(uint4*)(Bs + r * 32 + seg * 8) =
          *(const uint4*)(B + (size_t)(n0 + r) * K + k0 + seg * 8);
      if (k0 + 32 < K) {
        __builtin_prefetch(A + (size_t)(m0 + r) * K + k0 + 32 + seg * 8, 0, 0);
        __builtin_prefetch(B + (size_t)(n0 + r) * K + k0 + 32 + seg * 8, 0, 0);
      }
    }
#endif
    __syncthreads();
    FragAB a;
    const unsigned short* ap = As + (wm * 16 + lr) * 32 + lh * 16;
    a.q[0] = *(const uint4*)(ap);
    a.q[1] = *(const uint4*)(ap + 8);
    FragAB b0, b1;
    const unsigned short* bp0 = Bs + (wn * 32 + lr) * 32 + lh * 16;
    b0.q[0] = *(const uint4*)(bp0);
    b0.q[1] = *(const uint4*)(bp0 + 8);
    const unsigned short* bp1 = Bs + (wn * 32 + 16 + lr) * 32 + lh * 16;
    b1.q[0] = *(const uint4*)(bp1);
    b1.q[1] = *(const uint4*)(bp1 + 8);
    acc0 = wmma_bf16(a.v, b0.v, acc0);
    acc1 = wmma_bf16(a.v, b1.v, acc1);
    __syncthreads();
  }
  // C/D layout: VGPR r -> (M = r + (lane>=16 ? 8:0), N = lane%16)
  const int row_l = (lane >= 16) ? 8 : 0;
  const int colA = n0 + wn * 32 + lr;
  const int colB = colA + 16;
  float bA = 0.f, bB = 0.f;
  if (BIAS) {
    bA = bias1[colA] + bias2[colA];
    bB = bias1[colB] + bias2[colB];
  }
#pragma unroll
  for (int rr = 0; rr < 8; ++rr) {
    int row = m0 + wm * 16 + row_l + rr;
    float vA = acc0[rr] + bA, vB = acc1[rr] + bB;
    if (OUT_BF16) {
      Cb[(size_t)row * N + colA] = f2bf(vA);
      Cb[(size_t)row * N + colB] = f2bf(vB);
    } else {
      Cf[(size_t)row * N + colA] = vA;
      Cf[(size_t)row * N + colB] = vB;
    }
  }
}

// ---------------------------------------------------------------------------
// One LSTM timestep: gates = Gt + h_prev @ W_hh^T, then cell update.
// Grid: 16 blocks (64 gate-columns each); 8 waves = 4 gates x 2 row-halves.
// ---------------------------------------------------------------------------
__global__ void lstm_step_kernel(const unsigned short* __restrict__ Hprev,
                                 const unsigned short* __restrict__ Whh,
                                 const float* __restrict__ Gt,
                                 float* __restrict__ c_state,
                                 unsigned short* __restrict__ Hout) {
  __shared__ __align__(16) unsigned short hs[32 * 32];
  __shared__ __align__(16) unsigned short wsb[256 * 32];
  __shared__ float gbuf[4][32][64];
  const int tid = threadIdx.x;
  const int lane = tid & 31, w = tid >> 5;
  const int gate = w >> 1, mh = w & 1;
  const int j0 = blockIdx.x * 64;
  const int lr = lane & 15, lh = lane >> 4;
  v8f acc[4];
#pragma unroll
  for (int i = 0; i < 4; ++i) acc[i] = zero8();

  const int wrow = (tid >> 6) * NHID + j0 + (tid & 63);  // W_hh row for slot tid
  for (int k0 = 0; k0 < NHID; k0 += 32) {
    if (tid < 128) {
      int r = tid >> 2, seg = tid & 3;
      *(uint4*)(hs + r * 32 + seg * 8) =
          *(const uint4*)(Hprev + (size_t)r * NHID + k0 + seg * 8);
    }
#if HAVE_TDM
    if (w == 0) {
#pragma unroll
      for (int g = 0; g < 4; ++g) {
        tdm_load_2d_bf16(Whh + (size_t)(g * NHID + j0) * NHID + k0,
                         lds_off_u32(wsb + g * 64 * 32), 64, 32, NHID, 64);
      }
      tdm_wait0();
    }
#else
    {
      const uint4* src = (const uint4*)(Whh + (size_t)wrow * NHID + k0);
      uint4* dst = (uint4*)(wsb + tid * 32);
      dst[0] = src[0]; dst[1] = src[1]; dst[2] = src[2]; dst[3] = src[3];
      if (k0 + 32 < NHID)
        __builtin_prefetch(Whh + (size_t)wrow * NHID + k0 + 32, 0, 0);
    }
#endif
    __syncthreads();
    FragAB a;
    const unsigned short* ap = hs + (mh * 16 + lr) * 32 + lh * 16;
    a.q[0] = *(const uint4*)(ap);
    a.q[1] = *(const uint4*)(ap + 8);
#pragma unroll
    for (int nt = 0; nt < 4; ++nt) {
      FragAB b;
      const unsigned short* bp = wsb + (gate * 64 + nt * 16 + lr) * 32 + lh * 16;
      b.q[0] = *(const uint4*)(bp);
      b.q[1] = *(const uint4*)(bp + 8);
      acc[nt] = wmma_bf16(a.v, b.v, acc[nt]);
    }
    __syncthreads();
  }
  // stage pre-activations (+precomputed x-proj and biases) into LDS
  {
    int row_base = mh * 16 + ((lane >= 16) ? 8 : 0);
#pragma unroll
    for (int nt = 0; nt < 4; ++nt) {
      int col = nt * 16 + lr;
#pragma unroll
      for (int rr = 0; rr < 8; ++rr) {
        int row = row_base + rr;
        gbuf[gate][row][col] =
            acc[nt][rr] + Gt[(size_t)row * GDIM + gate * NHID + j0 + col];
      }
    }
  }
  __syncthreads();
  // LSTM cell (PyTorch gate order i,f,g,o)
  {
    int base = tid * 8;
#pragma unroll
    for (int e = 0; e < 8; ++e) {
      int idx = base + e;
      int row = idx >> 6, col = idx & 63;
      float gi = gbuf[0][row][col];
      float gf = gbuf[1][row][col];
      float gg = gbuf[2][row][col];
      float go = gbuf[3][row][col];
      int jg = j0 + col;
      float c_old = c_state[row * NHID + jg];
      float cn = sigm(gf) * c_old + sigm(gi) * tanhf(gg);
      float hn = sigm(go) * tanhf(cn);
      c_state[row * NHID + jg] = cn;
      Hout[(size_t)row * NHID + jg] = f2bf(hn);
    }
  }
}

// ---------------------------------------------------------------------------
// WMMA logits tile + streaming per-tile logsumexp partials + target capture.
// Tile: 64 rows x 128 cols. mode: 0=head, 1=tail0, 2=tail1.
// ---------------------------------------------------------------------------
__global__ void wmma_logits_lse_kernel(const unsigned short* __restrict__ A,
                                       const unsigned short* __restrict__ W,
                                       int N, int K,
                                       const int* __restrict__ targets, int mode,
                                       float* __restrict__ pMax,
                                       float* __restrict__ pSum, int ntiles,
                                       float* __restrict__ tgtLogit) {
  __shared__ __align__(16) unsigned short As[64 * 32];
  __shared__ __align__(16) unsigned short Bs[128 * 32];
  __shared__ float Lt[64][132];
  const int tid = threadIdx.x;
  const int lane = tid & 31, w = tid >> 5;
  const int wm = w & 3, wn = w >> 2;
  const int tileN = blockIdx.x;
  const int n0 = tileN * LTILE, m0 = blockIdx.y * 64;
  const int lr = lane & 15, lh = lane >> 4;
  v8f acc[4];
#pragma unroll
  for (int i = 0; i < 4; ++i) acc[i] = zero8();

  for (int k0 = 0; k0 < K; k0 += 32) {
#if HAVE_TDM
    if (w == 0) {
      tdm_load_2d_bf16(A + (size_t)m0 * K + k0, lds_off_u32(As), 64, 32, K, 64);
      // TDM zero-fills rows beyond (N - n0): boundary tiles handled for free.
      tdm_load_2d_bf16(W + (size_t)n0 * K + k0, lds_off_u32(Bs), 128, 32, K,
                       (unsigned)(N - n0));
      tdm_wait0();
    }
#else
    {
      int r = tid >> 2, seg = tid & 3;
      *(uint4*)(As + r * 32 + seg * 8) =
          *(const uint4*)(A + (size_t)(m0 + r) * K + k0 + seg * 8);
    }
    {
      int r = tid >> 1, seg = tid & 1;
      uint4 z0 = {0, 0, 0, 0}, z1 = {0, 0, 0, 0};
      if (n0 + r < N) {
        const uint4* src = (const uint4*)(W + (size_t)(n0 + r) * K + k0 + seg * 16);
        z0 = src[0];
        z1 = src[1];
        if (k0 + 32 < K)
          __builtin_prefetch(W + (size_t)(n0 + r) * K + k0 + 32 + seg * 16, 0, 0);
      }
      uint4* dst = (uint4*)(Bs + r * 32 + seg * 16);
      dst[0] = z0;
      dst[1] = z1;
    }
#endif
    __syncthreads();
    FragAB a;
    const unsigned short* ap = As + (wm * 16 + lr) * 32 + lh * 16;
    a.q[0] = *(const uint4*)(ap);
    a.q[1] = *(const uint4*)(ap + 8);
#pragma unroll
    for (int nt = 0; nt < 4; ++nt) {
      FragAB b;
      const unsigned short* bp = Bs + (wn * 64 + nt * 16 + lr) * 32 + lh * 16;
      b.q[0] = *(const uint4*)(bp);
      b.q[1] = *(const uint4*)(bp + 8);
      acc[nt] = wmma_bf16(a.v, b.v, acc[nt]);
    }
    __syncthreads();
  }
  // scatter logits tile to LDS
  {
    int row_base = wm * 16 + ((lane >= 16) ? 8 : 0);
#pragma unroll
    for (int nt = 0; nt < 4; ++nt) {
      int col = wn * 64 + nt * 16 + lr;
#pragma unroll
      for (int rr = 0; rr < 8; ++rr) Lt[row_base + rr][col] = acc[nt][rr];
    }
  }
  __syncthreads();
  if (tid < 64) {
    int row = tid, rowG = m0 + row;
    int nv = N - n0;
    if (nv > LTILE) nv = LTILE;
    float mx = -3.0e38f;
    for (int c = 0; c < nv; ++c) mx = fmaxf(mx, Lt[row][c]);
    float s = 0.f;
    for (int c = 0; c < nv; ++c) s += __expf(Lt[row][c] - mx);
    pMax[(size_t)rowG * ntiles + tileN] = mx;
    pSum[(size_t)rowG * ntiles + tileN] = s;
    int t = targets[rowG];
    int gi;
    if (mode == 0) {
      int cl = (t >= CUT0) + (t >= CUT1);
      gi = (cl == 0) ? t : (SL + cl - 1);
    } else {
      int lo = (mode == 1) ? CUT0 : CUT1;
      gi = t - lo;
      if (gi < 0) gi = 0;
      if (gi > N - 1) gi = N - 1;
    }
    if (gi >= n0 && gi < n0 + nv) tgtLogit[rowG] = Lt[row][gi - n0];
  }
}

// ---------------------------------------------------------------------------
// Merge per-tile (max,sumexp) partials -> output[row]
// ---------------------------------------------------------------------------
__global__ void finalize_kernel(const float* __restrict__ pmH,
                                const float* __restrict__ psH,
                                const float* __restrict__ tgtH,
                                const float* __restrict__ pm0,
                                const float* __restrict__ ps0,
                                const float* __restrict__ tgt0,
                                const float* __restrict__ pm1,
                                const float* __restrict__ ps1,
                                const float* __restrict__ tgt1,
                                const int* __restrict__ targets,
                                float* __restrict__ out) {
  int row = blockIdx.x * blockDim.x + threadIdx.x;
  if (row >= NROWS) return;
  auto lse = [](const float* pm, const float* ps, int nt, int r) -> float {
    float M = -3.0e38f;
    for (int i = 0; i < nt; ++i) M = fmaxf(M, pm[(size_t)r * nt + i]);
    float S = 0.f;
    for (int i = 0; i < nt; ++i)
      S += ps[(size_t)r * nt + i] * __expf(pm[(size_t)r * nt + i] - M);
    return M + __logf(S);
  };
  float o = tgtH[row] - lse(pmH, psH, NT_HEAD, row);
  int t = targets[row];
  int cl = (t >= CUT0) + (t >= CUT1);
  if (cl == 1) o += tgt0[row] - lse(pm0, ps0, NT_T0, row);
  else if (cl == 2) o += tgt1[row] - lse(pm1, ps1, NT_T1, row);
  out[row] = o;
}

__global__ void loss_kernel(const float* __restrict__ out,
                            float* __restrict__ loss_out) {
  __shared__ float red[256];
  float s = 0.f;
  for (int i = threadIdx.x; i < NROWS; i += 256) s += out[i];
  red[threadIdx.x] = s;
  __syncthreads();
  for (int st = 128; st > 0; st >>= 1) {
    if (threadIdx.x < st) red[threadIdx.x] += red[threadIdx.x + st];
    __syncthreads();
  }
  if (threadIdx.x == 0) *loss_out = -red[0] / (float)NROWS;
}

// ---------------------------------------------------------------------------
extern "C" void kernel_launch(void* const* d_in, const int* in_sizes, int n_in,
                              void* d_out, int out_size, void* d_ws,
                              size_t ws_size, hipStream_t stream) {
  const int*   tokens  = (const int*)d_in[0];
  const int*   targets = (const int*)d_in[1];
  const float* emb_W   = (const float*)d_in[2];
  const float* w_ih0   = (const float*)d_in[3];
  const float* w_hh0   = (const float*)d_in[4];
  const float* b_ih0   = (const float*)d_in[5];
  const float* b_hh0   = (const float*)d_in[6];
  const float* w_ih1   = (const float*)d_in[7];
  const float* w_hh1   = (const float*)d_in[8];
  const float* b_ih1   = (const float*)d_in[9];
  const float* b_hh1   = (const float*)d_in[10];
  const float* h0      = (const float*)d_in[11];
  const float* c0      = (const float*)d_in[12];
  const float* head_W  = (const float*)d_in[13];
  const float* t0w1    = (const float*)d_in[14];
  const float* t0w2    = (const float*)d_in[15];
  const float* t1w1    = (const float*)d_in[16];
  const float* t1w2    = (const float*)d_in[17];
  float* out = (float*)d_out;
  (void)in_sizes; (void)n_in; (void)ws_size;

  char* wp = (char*)d_ws;
  auto alloc = [&](size_t bytes) -> void* {
    void* p = (void*)wp;
    wp += (bytes + 255) & ~(size_t)255;
    return p;
  };
  unsigned short* Xb     = (unsigned short*)alloc((size_t)NROWS * NINP * 2);
  unsigned short* Wih0b  = (unsigned short*)alloc((size_t)GDIM * NINP * 2);
  unsigned short* Whh0b  = (unsigned short*)alloc((size_t)GDIM * NHID * 2);
  unsigned short* Wih1b  = (unsigned short*)alloc((size_t)GDIM * NHID * 2);
  unsigned short* Whh1b  = (unsigned short*)alloc((size_t)GDIM * NHID * 2);
  unsigned short* HeadWb = (unsigned short*)alloc((size_t)HEADN * NHID * 2);
  unsigned short* T0w1b  = (unsigned short*)alloc((size_t)T0HS * NHID * 2);
  unsigned short* T0w2b  = (unsigned short*)alloc((size_t)T0OS * T0HS * 2);
  unsigned short* T1w1b  = (unsigned short*)alloc((size_t)T1HS * NHID * 2);
  unsigned short* T1w2b  = (unsigned short*)alloc((size_t)T1OS * T1HS * 2);
  float*          G      = (float*)alloc((size_t)NROWS * GDIM * 4);  // L0/L1
  unsigned short* H0buf  = (unsigned short*)alloc((size_t)(NROWS + BATCH) * NHID * 2);
  unsigned short* H1buf  = (unsigned short*)alloc((size_t)(NROWS + BATCH) * NHID * 2);
  float*          c_st   = (float*)alloc((size_t)BATCH * NHID * 4);
  unsigned short* T0Hb   = (unsigned short*)alloc((size_t)NROWS * T0HS * 2);
  unsigned short* T1Hb   = (unsigned short*)alloc((size_t)NROWS * T1HS * 2);
  float* pmH  = (float*)alloc((size_t)NROWS * NT_HEAD * 4);
  float* psH  = (float*)alloc((size_t)NROWS * NT_HEAD * 4);
  float* pm0  = (float*)alloc((size_t)NROWS * NT_T0 * 4);
  float* ps0  = (float*)alloc((size_t)NROWS * NT_T0 * 4);
  float* pm1  = (float*)alloc((size_t)NROWS * NT_T1 * 4);
  float* ps1  = (float*)alloc((size_t)NROWS * NT_T1 * 4);
  float* tgtH = (float*)alloc((size_t)NROWS * 4);
  float* tgt0 = (float*)alloc((size_t)NROWS * 4);
  float* tgt1 = (float*)alloc((size_t)NROWS * 4);

  auto cast = [&](const float* src, unsigned short* dst, long n) {
    cast_f32_bf16_kernel<<<dim3((unsigned)((n / 4 + 255) / 256)), dim3(256), 0,
                           stream>>>(src, dst, n);
  };
  // 1) weight casts
  cast(w_ih0, Wih0b, (long)GDIM * NINP);
  cast(w_hh0, Whh0b, (long)GDIM * NHID);
  cast(w_ih1, Wih1b, (long)GDIM * NHID);
  cast(w_hh1, Whh1b, (long)GDIM * NHID);
  cast(head_W, HeadWb, (long)HEADN * NHID);
  cast(t0w1, T0w1b, (long)T0HS * NHID);
  cast(t0w2, T0w2b, (long)T0OS * T0HS);
  cast(t1w1, T1w1b, (long)T1HS * NHID);
  cast(t1w2, T1w2b, (long)T1OS * T1HS);
  // 2) embedding gather
  embed_kernel<<<dim3(NROWS), dim3(256), 0, stream>>>(tokens, emb_W, Xb);

  // ---- layer 0 ----
  wmma_gemm_kernel<false, true><<<dim3(GDIM / 64, NROWS / 64), dim3(256), 0,
                                  stream>>>(Xb, Wih0b, G, nullptr, b_ih0, b_hh0,
                                            NROWS, GDIM, NINP);
  cast(h0 + 0 * BATCH * NHID, H0buf, (long)BATCH * NHID);
  hipMemcpyAsync(c_st, c0 + 0 * BATCH * NHID, (size_t)BATCH * NHID * 4,
                 hipMemcpyDeviceToDevice, stream);
  for (int t = 0; t < SEQ; ++t) {
    lstm_step_kernel<<<dim3(NHID / 64), dim3(256), 0, stream>>>(
        H0buf + (size_t)t * BATCH * NHID, Whh0b, G + (size_t)t * BATCH * GDIM,
        c_st, H0buf + (size_t)(t + 1) * BATCH * NHID);
  }
  unsigned short* H0 = H0buf + (size_t)BATCH * NHID;  // [4096,1024] bf16

  // ---- layer 1 ----
  wmma_gemm_kernel<false, true><<<dim3(GDIM / 64, NROWS / 64), dim3(256), 0,
                                  stream>>>(H0, Wih1b, G, nullptr, b_ih1, b_hh1,
                                            NROWS, GDIM, NHID);
  cast(h0 + 1 * BATCH * NHID, H1buf, (long)BATCH * NHID);
  hipMemcpyAsync(c_st, c0 + 1 * BATCH * NHID, (size_t)BATCH * NHID * 4,
                 hipMemcpyDeviceToDevice, stream);
  for (int t = 0; t < SEQ; ++t) {
    lstm_step_kernel<<<dim3(NHID / 64), dim3(256), 0, stream>>>(
        H1buf + (size_t)t * BATCH * NHID, Whh1b, G + (size_t)t * BATCH * GDIM,
        c_st, H1buf + (size_t)(t + 1) * BATCH * NHID);
  }
  unsigned short* Hfin = H1buf + (size_t)BATCH * NHID;  // final hidden bf16

  // ---- adaptive softmax: head ----
  wmma_logits_lse_kernel<<<dim3(NT_HEAD, NROWS / 64), dim3(256), 0, stream>>>(
      Hfin, HeadWb, HEADN, NHID, targets, 0, pmH, psH, NT_HEAD, tgtH);
  // ---- tail 0 ----
  wmma_gemm_kernel<true, false><<<dim3(T0HS / 64, NROWS / 64), dim3(256), 0,
                                  stream>>>(Hfin, T0w1b, nullptr, T0Hb, nullptr,
                                            nullptr, NROWS, T0HS, NHID);
  wmma_logits_lse_kernel<<<dim3(NT_T0, NROWS / 64), dim3(256), 0, stream>>>(
      T0Hb, T0w2b, T0OS, T0HS, targets, 1, pm0, ps0, NT_T0, tgt0);
  // ---- tail 1 ----
  wmma_gemm_kernel<true, false><<<dim3(T1HS / 64, NROWS / 64), dim3(256), 0,
                                  stream>>>(Hfin, T1w1b, nullptr, T1Hb, nullptr,
                                            nullptr, NROWS, T1HS, NHID);
  wmma_logits_lse_kernel<<<dim3(NT_T1, NROWS / 64), dim3(256), 0, stream>>>(
      T1Hb, T1w2b, T1OS, T1HS, targets, 2, pm1, ps1, NT_T1, tgt1);

  // ---- finalize + loss ----
  finalize_kernel<<<dim3(NROWS / 256), dim3(256), 0, stream>>>(
      pmH, psH, tgtH, pm0, ps0, tgt0, pm1, ps1, tgt1, targets, out);
  if (out_size > NROWS)
    loss_kernel<<<dim3(1), dim3(256), 0, stream>>>(out, out + NROWS);
}
